// GrelPose_13254269075987
// MI455X (gfx1250) — matrix operand order repaired and checked by reading, not compiled, not run on hardware
//
#include <hip/hip_runtime.h>
#include <hip/hip_bf16.h>
#include <math.h>
#include <stdint.h>

typedef _Float16 f16;
typedef __attribute__((ext_vector_type(16))) _Float16 v16h;
typedef __attribute__((ext_vector_type(8)))  _Float16 v8h;
typedef __attribute__((ext_vector_type(8)))  float    v8f;
typedef unsigned int u32;
typedef __attribute__((ext_vector_type(4))) u32 v4u;
typedef __attribute__((ext_vector_type(4))) int v4i;
typedef __attribute__((ext_vector_type(8))) int v8i;

#define BB 32
#define DD 256
#define HH 32
#define WW 42
#define HWN 1344            // 32*42
#define NT_CORR 84          // 1344/16
#define CP1 544             // 517 padded to mult of 32
#define H1 16
#define W1 21
#define HW1 336
#define C1 256
#define H2 8
#define W2 11
#define HW2 88
#define C2 128

// ------------------------------------------------ CDNA5 feature detection
#if defined(__has_builtin)
#  if __has_builtin(__builtin_amdgcn_tensor_load_to_lds)
#    define USE_TDM 1
#  else
#    define USE_TDM 0
#  endif
#  if __has_builtin(__builtin_amdgcn_s_wait_tensorcnt)
#    define WAIT_TENSOR(n) __builtin_amdgcn_s_wait_tensorcnt(n)
#  else
#    define WAIT_TENSOR(n) asm volatile("s_wait_tensorcnt %0" ::"i"(n))
#  endif
#else
#  define USE_TDM 0
#  define WAIT_TENSOR(n)
#endif

// ---------------------------------------------------------------- fragments
__device__ __forceinline__ v16h frag_a(const f16* __restrict__ rowBase, int kb, int lane) {
  // 16-bit A 16x32: lane<16 -> M=lane, K in kb+0..15; lane>=16 -> K kb+16..31
  int khalf = (lane >> 4) * 16;
  v8h lo = *(const v8h*)(rowBase + kb + khalf);
  v8h hi = *(const v8h*)(rowBase + kb + khalf + 8);
  union { v16h v; v8h h[2]; } u;
  u.h[0] = lo; u.h[1] = hi;
  return u.v;
}

__device__ __forceinline__ v16h frag_b(const f16* __restrict__ colBase, int kb, int lane) {
  // 16-bit B 32x16: lane&15 = N column, lane>=16 holds K rows 16..31
  int kbase = (lane >> 4) * 16;
  return *(const v16h*)(colBase + kb + kbase);
}

__device__ __forceinline__ v8f wmma_f16(v16h a, v16h b, v8f c) {
  return __builtin_amdgcn_wmma_f32_16x16x32_f16(false, a, false, b, (short)0, c, false, false);
}

#if USE_TDM
// Issue a TDM 2D tile load: 16 rows x 256 bytes, row stride 512 bytes, into LDS.
// data_size = 8B units: tile_dim0 = 32 units (256B), tile_dim1 = 16 rows, stride0 = 64 units.
__device__ __forceinline__ void tdm_load_tile(const f16* gsrc, u32 lds_byte_off) {
  unsigned long long ga = (unsigned long long)(uintptr_t)gsrc;
  v4u g0;
  g0.x = 1u;                                        // count=1 (valid user descriptor)
  g0.y = lds_byte_off;                              // lds_addr
  g0.z = (u32)ga;                                   // global_addr[31:0]
  g0.w = (u32)((ga >> 32) & 0x01ffffffu) | (2u << 30);  // global_addr[56:32] | type=2
  v8i g1;
  g1[0] = (int)(3u << 16);                          // data_size = 8B
  g1[1] = 0;                                        // tensor_dim0[15:0]=0 (dim0 = 1<<20)
  g1[2] = 16;                                       // tensor_dim0[31:16]=16, tensor_dim1[15:0]=0
  g1[3] = (int)(16u | (32u << 16));                 // tensor_dim1[31:16]=16, tile_dim0=32
  g1[4] = 16;                                       // tile_dim1=16, tile_dim2=0
  g1[5] = 64;                                       // tensor_dim0_stride = 64 units (512B)
  g1[6] = 0;
  g1[7] = 0;
  v4i z4 = {0, 0, 0, 0};
#if defined(__clang_major__) && (__clang_major__ >= 23)
  v8i z8 = {0, 0, 0, 0, 0, 0, 0, 0};
  __builtin_amdgcn_tensor_load_to_lds(g0, g1, z4, z4, z8, 0);
#else
  __builtin_amdgcn_tensor_load_to_lds(g0, g1, z4, z4, 0);
#endif
}
#endif

// ---------------------------------------------------------------- normalize & pack f16 [b][pixel][d]
__global__ void k_normalize(const float* __restrict__ f1, const float* __restrict__ f2,
                            f16* __restrict__ A16, f16* __restrict__ B16) {
  int idx = blockIdx.x * blockDim.x + threadIdx.x;
  int total = BB * HWN;
  if (idx >= 2 * total) return;
  int which = idx / total;
  int r = idx - which * total;
  int b = r / HWN, m = r - b * HWN;
  const float* src = which ? f2 : f1;
  f16* dst = which ? B16 : A16;
  float ss = 0.f;
  for (int d = 0; d < DD; d++) {
    float v = src[((size_t)b * DD + d) * HWN + m];
    ss += v * v;
  }
  float inv = 1.0f / fmaxf(sqrtf(ss), 1e-12f);
  f16* o = dst + ((size_t)b * HWN + m) * DD;
  for (int d = 0; d < DD; d++)
    o[d] = (f16)(src[((size_t)b * DD + d) * HWN + m] * inv);
}

// ---------------------------------------------------------------- fused WMMA correlation + online softmax max/argmax
// TDM path: B tiles streamed into LDS (double-buffered, half-K chunks of 16x128 f16).
// A fragments are re-loaded from global per chunk (L2-hot, avoids VGPR spill to scratch).
__global__ __launch_bounds__(128) void k_corr(const f16* __restrict__ A16,
                                              const f16* __restrict__ B16,
                                              int* __restrict__ inds,
                                              float* __restrict__ val) {
  int b    = blockIdx.y;
  int mt   = blockIdx.x;
  int wave = threadIdx.x >> 5;
  int lane = threadIdx.x & 31;

  const f16* Abase = A16 + ((size_t)b * HWN + mt * 16 + (lane & 15)) * DD;

  float rm[8], rs[8]; int ra[8];
#pragma unroll
  for (int r = 0; r < 8; r++) { rm[r] = -3.0e38f; rs[r] = 0.f; ra[r] = 0; }

  const v8f vzero = {};

#if USE_TDM
  __shared__ f16 btile[4][2][16 * 128];   // 4 waves x double buffer x 4KB chunk
  u32 lds_base = (u32)(uintptr_t)&btile[wave][0][0];

  const int NCHUNK = NT_CORR / 4 * 2;     // 21 tiles * 2 half-K chunks
  // prime the pipeline with chunk 0 (tile nt=wave, k-half 0)
  tdm_load_tile(B16 + ((size_t)b * HWN + wave * 16) * DD, lds_base);

  v8f acc = vzero;
  int p = 0;
  for (int ci = 0; ci < NCHUNK; ci++) {
    int nt = wave + 4 * (ci >> 1);
    int h  = ci & 1;
    bool more = (ci + 1 < NCHUNK);
    if (more) {
      int nt2 = wave + 4 * ((ci + 1) >> 1);
      int h2  = (ci + 1) & 1;
      tdm_load_tile(B16 + ((size_t)b * HWN + nt2 * 16) * DD + h2 * 128,
                    lds_base + (u32)((p ^ 1) * 16 * 128 * 2));
    }
    // A fragments for this half (global loads overlap the tensor DMA wait)
    v16h a0 = frag_a(Abase, h * 128 +  0, lane);
    v16h a1 = frag_a(Abase, h * 128 + 32, lane);
    v16h a2 = frag_a(Abase, h * 128 + 64, lane);
    v16h a3 = frag_a(Abase, h * 128 + 96, lane);
    if (more) { WAIT_TENSOR(1); } else { WAIT_TENSOR(0); }
    const f16* bl = &btile[wave][p][(lane & 15) * 128 + (lane >> 4) * 16];
    acc = wmma_f16(a0, *(const v16h*)(bl +  0), acc);
    acc = wmma_f16(a1, *(const v16h*)(bl + 32), acc);
    acc = wmma_f16(a2, *(const v16h*)(bl + 64), acc);
    acc = wmma_f16(a3, *(const v16h*)(bl + 96), acc);
    p ^= 1;
    if (h == 1) {
#else
  // hoist all A fragments (16 rows x 256 k) into registers
  v16h af[8];
#pragma unroll
  for (int kk = 0; kk < 8; kk++) af[kk] = frag_a(Abase, kk * 32, lane);
  for (int nt = wave; nt < NT_CORR; nt += 4) {
    const f16* Bbase = B16 + ((size_t)b * HWN + nt * 16 + (lane & 15)) * DD;
    // software pipeline: issue all 8 B-fragment loads before consuming
    v16h bfr[8];
#pragma unroll
    for (int kk = 0; kk < 8; kk++) bfr[kk] = frag_b(Bbase, kk * 32, lane);
    v8f acc = vzero;
#pragma unroll
    for (int kk = 0; kk < 8; kk++) acc = wmma_f16(af[kk], bfr[kk], acc);
    {
#endif
      // ---- online softmax merge for this 16x16 tile ----
#pragma unroll
      for (int r = 0; r < 8; r++) {
        float v = acc[r];
        int   a = nt * 16 + (lane & 15);
        float m = v; int ai = a;
#pragma unroll
        for (int mask = 1; mask < 16; mask <<= 1) {
          float om = __shfl_xor(m, mask, 32);
          int   oa = __shfl_xor(ai, mask, 32);
          if (om > m || (om == m && oa < ai)) { m = om; ai = oa; }
        }
        float e = __expf(v - m);
#pragma unroll
        for (int mask = 1; mask < 16; mask <<= 1)
          e += __shfl_xor(e, mask, 32);
        float nm = fmaxf(rm[r], m);
        rs[r] = rs[r] * __expf(rm[r] - nm) + e * __expf(m - nm);
        if (m > rm[r]) ra[r] = ai;
        rm[r] = nm;
      }
#if USE_TDM
      acc = vzero;
#endif
    }
  }

  __shared__ float smax[4][16];
  __shared__ float ssum[4][16];
  __shared__ int   sarg[4][16];
  if ((lane & 15) == 0) {
    int half = lane >> 4;  // C layout: vgpr r, lanes>=16 -> rows 8..15
#pragma unroll
    for (int r = 0; r < 8; r++) {
      smax[wave][half * 8 + r] = rm[r];
      ssum[wave][half * 8 + r] = rs[r];
      sarg[wave][half * 8 + r] = ra[r];
    }
  }
  __syncthreads();
  int tid = threadIdx.x;
  if (tid < 16) {
    float gm = smax[0][tid]; float gs = ssum[0][tid]; int ga = sarg[0][tid];
    for (int w = 1; w < 4; w++) {
      float m = smax[w][tid]; float s = ssum[w][tid]; int a = sarg[w][tid];
      float nm = fmaxf(gm, m);
      gs = gs * __expf(gm - nm) + s * __expf(m - nm);
      if (m > gm || (m == gm && a < ga)) ga = a;
      gm = nm;
    }
    int mrow = mt * 16 + tid;
    inds[(size_t)b * HWN + mrow] = ga;
    val [(size_t)b * HWN + mrow] = 1.0f / gs;   // max(softmax) = 1/sum(exp(x-max))
  }
}

// ---------------------------------------------------------------- build relu(bn1(feat_agg)) as f16 NHWC[CP1]
__global__ void k_aggregate(const float* __restrict__ f1, const float* __restrict__ f2,
                            const int* __restrict__ inds, const float* __restrict__ val,
                            const float* __restrict__ g, const float* __restrict__ bb,
                            f16* __restrict__ y1) {
  int idx = blockIdx.x * blockDim.x + threadIdx.x;
  if (idx >= BB * HWN) return;
  int b = idx / HWN, m = idx - b * HWN;
  int ind = inds[(size_t)b * HWN + m];
  float vm = val[(size_t)b * HWN + m];
  int ox = m % WW, oy = m / WW;
  f16* out = y1 + ((size_t)b * HWN + m) * CP1;
  const float s = rsqrtf(1.0f + 1e-5f);
  for (int c = 0; c < 517; c++) {
    float x;
    if      (c < 256)  x = f1[((size_t)b * DD + c) * HWN + m];
    else if (c == 256) x = (float)ox;
    else if (c == 257) x = (float)oy;
    else if (c < 514)  x = f2[((size_t)b * DD + (c - 258)) * HWN + ind];
    else if (c == 514) x = (float)(ind % WW);
    else if (c == 515) x = (float)(ind / WW);
    else               x = vm;
    float v = x * (g[c] * s) + bb[c];
    out[c] = (f16)fmaxf(v, 0.f);
  }
  for (int c = 517; c < CP1; c++) out[c] = (f16)0.f;
}

// ---------------------------------------------------------------- pack OIHW f32 weights -> f16 [O][taps*Cpad]
__global__ void k_packw(const float* __restrict__ w, f16* __restrict__ wp,
                        int O, int Cin, int Cpad, int taps) {
  int Ktot = taps * Cpad;
  int i = blockIdx.x * blockDim.x + threadIdx.x;
  if (i >= O * Ktot) return;
  int o = i / Ktot, k = i - o * Ktot;
  int tap = k / Cpad, c = k - tap * Cpad;
  f16 v = (f16)0.f;
  if (c < Cin) v = (f16)w[((size_t)o * Cin + c) * taps + tap];
  wp[i] = v;
}

// ---------------------------------------------------------------- implicit-GEMM WMMA conv (NHWC f16 in, NHWC f32 out)
__global__ __launch_bounds__(128) void k_conv(const f16* __restrict__ yin, int Hin, int Win, int Cpad,
                                              const f16* __restrict__ wp, int Cout,
                                              int Hout, int Wout, int stride, int pad,
                                              int KW, int taps,
                                              const float* __restrict__ addsrc,
                                              const float* __restrict__ bias,
                                              float* __restrict__ out) {
  int wave = threadIdx.x >> 5, lane = threadIdx.x & 31;
  int mt = blockIdx.x;
  int nt = blockIdx.y * 4 + wave;
  if (nt * 16 >= Cout) return;
  int Ktot = taps * Cpad;
  int hw = Hout * Wout;
  int m  = mt * 16 + (lane & 15);
  int b  = m / hw;
  int r0 = m - b * hw;
  int oy = r0 / Wout, ox = r0 - oy * Wout;
  int iy0 = oy * stride - pad, ix0 = ox * stride - pad;
  const f16* wbase = wp + (size_t)(nt * 16 + (lane & 15)) * Ktot;

  auto loadA = [&](int kb) -> v16h {
    int tap = kb / Cpad;             // Cpad multiple of 32: chunk never crosses a tap
    int c0  = kb - tap * Cpad;
    int ky  = tap / KW, kx = tap - ky * KW;
    int iy  = iy0 + ky, ix = ix0 + kx;
    v16h a = {};
    if ((unsigned)iy < (unsigned)Hin && (unsigned)ix < (unsigned)Win) {
      const f16* rowBase = yin + (((size_t)b * Hin + iy) * Win + ix) * Cpad + c0;
      a = frag_a(rowBase, 0, lane);
    }
    return a;
  };

  // register double-buffered k pipeline
  v8f acc = {};
  v16h a_cur = loadA(0);
  v16h b_cur = frag_b(wbase, 0, lane);
  for (int kb = 0; kb < Ktot; kb += 32) {
    v16h a_nxt = {}, b_nxt = {};
    if (kb + 32 < Ktot) {
      __builtin_prefetch(wbase + kb + 160, 0, 1);   // global_prefetch on weight stream
      a_nxt = loadA(kb + 32);
      b_nxt = frag_b(wbase, kb + 32, lane);
    }
    acc = wmma_f16(a_cur, b_cur, acc);
    a_cur = a_nxt; b_cur = b_nxt;
  }

  int col = nt * 16 + (lane & 15);
#pragma unroll
  for (int r = 0; r < 8; r++) {
    int mm = mt * 16 + r + ((lane >> 4) * 8);    // C layout: vgpr r, halves -> rows r / r+8
    int b2 = mm / hw;
    int r2 = mm - b2 * hw;
    size_t o = ((size_t)b2 * hw + r2) * Cout + col;
    float v = acc[r];
    if (addsrc) v += addsrc[o];
    if (bias)   v += bias[col];
    out[o] = v;
  }
}

// ---------------------------------------------------------------- relu(bn(x)) f32 NHWC -> f16 NHWC padded
__global__ void k_bnrelu(const float* __restrict__ in, const float* __restrict__ g,
                         const float* __restrict__ bb, f16* __restrict__ out,
                         int Npix, int C, int Cpad) {
  int i = blockIdx.x * blockDim.x + threadIdx.x;
  if (i >= Npix * Cpad) return;
  int p = i / Cpad, c = i - p * Cpad;
  f16 v = (f16)0.f;
  if (c < C) {
    float s = rsqrtf(1.0f + 1e-5f);
    float x = in[(size_t)p * C + c] * (g[c] * s) + bb[c];
    v = (f16)fmaxf(x, 0.f);
  }
  out[i] = v;
}

// ---------------------------------------------------------------- head MLP layer 1 (1408 -> 128, relu)
__global__ void k_head1(const float* __restrict__ c3out, const float* __restrict__ w1,
                        const float* __restrict__ b1v, float* __restrict__ hdd) {
  int i = blockIdx.x * blockDim.x + threadIdx.x;
  if (i >= BB * 128) return;
  int b = i / 128, j = i - b * 128;
  float s = 0.f;
  for (int k = 0; k < 1408; k++) {
    int c = k / HW2, hw = k - c * HW2;                 // v = x.reshape: c-major over (c,h,w)
    s += c3out[((size_t)b * HW2 + hw) * 16 + c] * w1[(size_t)j * 1408 + k];
  }
  hdd[i] = fmaxf(s + b1v[j], 0.f);
}

// ---------------------------------------------------------------- head layer 2 + quaternion -> (R, t)
__global__ void k_head2(const float* __restrict__ hdd, const float* __restrict__ w2,
                        const float* __restrict__ b2v, float* __restrict__ out) {
  int b = blockIdx.x * blockDim.x + threadIdx.x;
  if (b >= BB) return;
  float p[7];
  for (int j = 0; j < 7; j++) {
    float s = b2v[j];
    for (int k = 0; k < 128; k++) s += hdd[(size_t)b * 128 + k] * w2[(size_t)j * 128 + k];
    p[j] = s;
  }
  float r = p[0], ii = p[1], jj = p[2], kk = p[3];
  float two_s = 2.0f / (r * r + ii * ii + jj * jj + kk * kk);
  float* R = out + (size_t)b * 9;
  R[0] = 1.0f - two_s * (jj * jj + kk * kk);
  R[1] = two_s * (ii * jj - kk * r);
  R[2] = two_s * (ii * kk + jj * r);
  R[3] = two_s * (ii * jj + kk * r);
  R[4] = 1.0f - two_s * (ii * ii + kk * kk);
  R[5] = two_s * (jj * kk - ii * r);
  R[6] = two_s * (ii * kk - jj * r);
  R[7] = two_s * (jj * kk + ii * r);
  R[8] = 1.0f - two_s * (ii * ii + jj * jj);
  out[288 + (size_t)b * 3 + 0] = p[4];
  out[288 + (size_t)b * 3 + 1] = p[5];
  out[288 + (size_t)b * 3 + 2] = p[6];
}

// ================================================================ launcher
extern "C" void kernel_launch(void* const* d_in, const int* in_sizes, int n_in,
                              void* d_out, int out_size, void* d_ws, size_t ws_size,
                              hipStream_t stream) {
  const float* feat1    = (const float*)d_in[0];
  const float* feat2    = (const float*)d_in[1];
  const float* b1_bn1_g = (const float*)d_in[2];
  const float* b1_bn1_b = (const float*)d_in[3];
  const float* b1_c1_w  = (const float*)d_in[4];
  const float* b1_bn2_g = (const float*)d_in[5];
  const float* b1_bn2_b = (const float*)d_in[6];
  const float* b1_c2_w  = (const float*)d_in[7];
  const float* b1_sc_w  = (const float*)d_in[8];
  const float* b2_bn1_g = (const float*)d_in[9];
  const float* b2_bn1_b = (const float*)d_in[10];
  const float* b2_c1_w  = (const float*)d_in[11];
  const float* b2_bn2_g = (const float*)d_in[12];
  const float* b2_bn2_b = (const float*)d_in[13];
  const float* b2_c2_w  = (const float*)d_in[14];
  const float* b2_sc_w  = (const float*)d_in[15];
  const float* bn3_g    = (const float*)d_in[16];
  const float* bn3_b    = (const float*)d_in[17];
  const float* conv3_w  = (const float*)d_in[18];
  const float* conv3_b  = (const float*)d_in[19];
  const float* head_w1  = (const float*)d_in[20];
  const float* head_b1  = (const float*)d_in[21];
  const float* head_w2  = (const float*)d_in[22];
  const float* head_b2  = (const float*)d_in[23];
  float* out = (float*)d_out;

  size_t off = 0;
  auto alloc = [&](size_t bytes) -> void* {
    void* p = (char*)d_ws + off;
    off += (bytes + 255) & ~(size_t)255;
    return p;
  };

  f16*   A16    = (f16*)  alloc((size_t)BB * HWN * DD * 2);
  f16*   B16c   = (f16*)  alloc((size_t)BB * HWN * DD * 2);
  int*   inds   = (int*)  alloc((size_t)BB * HWN * 4);
  float* val    = (float*)alloc((size_t)BB * HWN * 4);
  f16*   y1     = (f16*)  alloc((size_t)BB * HWN * CP1 * 2);
  f16*   wp_c1  = (f16*)  alloc((size_t)C1 * 9 * CP1 * 2);
  f16*   wp_sc1 = (f16*)  alloc((size_t)C1 * 1 * CP1 * 2);
  f16*   wp_c2  = (f16*)  alloc((size_t)C1 * 9 * C1 * 2);
  f16*   wp_b2c1= (f16*)  alloc((size_t)C2 * 9 * C1 * 2);
  f16*   wp_b2sc= (f16*)  alloc((size_t)C2 * 1 * C1 * 2);
  f16*   wp_b2c2= (f16*)  alloc((size_t)C2 * 9 * C2 * 2);
  f16*   wp_c3  = (f16*)  alloc((size_t)16 * 1 * C2 * 2);
  float* sc1out = (float*)alloc((size_t)BB * HW1 * C1 * 4);
  float* c1out  = (float*)alloc((size_t)BB * HW1 * C1 * 4);
  f16*   z1     = (f16*)  alloc((size_t)BB * HW1 * C1 * 2);
  float* blk1   = (float*)alloc((size_t)BB * HW1 * C1 * 4);
  f16*   y2     = (f16*)  alloc((size_t)BB * HW1 * C1 * 2);
  float* sc2out = (float*)alloc((size_t)BB * HW2 * C2 * 4);
  float* c2out  = (float*)alloc((size_t)BB * HW2 * C2 * 4);
  f16*   z2     = (f16*)  alloc((size_t)BB * HW2 * C2 * 2);
  float* blk2   = (float*)alloc((size_t)BB * HW2 * C2 * 4);
  f16*   y3     = (f16*)  alloc((size_t)BB * HW2 * C2 * 2);
  float* c3out  = (float*)alloc((size_t)BB * HW2 * 16 * 4);
  float* hdd    = (float*)alloc((size_t)BB * 128 * 4);
  (void)ws_size;

  // 1) normalize both feature volumes, pack f16 [b][pixel][d]
  {
    int n = 2 * BB * HWN;
    k_normalize<<<(n + 255) / 256, 256, 0, stream>>>(feat1, feat2, A16, B16c);
  }
  // 2) fused WMMA correlation + online softmax argmax / 1-over-sumexp (TDM-staged B tiles)
  k_corr<<<dim3(NT_CORR, BB), 128, 0, stream>>>(A16, B16c, inds, val);
  // 3) warp + aggregate + bn1 + relu -> y1 (f16 NHWC, 544ch)
  {
    int n = BB * HWN;
    k_aggregate<<<(n + 255) / 256, 256, 0, stream>>>(feat1, feat2, inds, val, b1_bn1_g, b1_bn1_b, y1);
  }
  // 4) weight packing (f16 [Cout][taps*Cpad])
  auto packw = [&](const float* w, f16* wp, int O, int Cin, int Cpad, int taps) {
    int n = O * taps * Cpad;
    k_packw<<<(n + 255) / 256, 256, 0, stream>>>(w, wp, O, Cin, Cpad, taps);
  };
  packw(b1_c1_w, wp_c1,  C1, 517, CP1, 9);
  packw(b1_sc_w, wp_sc1, C1, 517, CP1, 1);
  packw(b1_c2_w, wp_c2,  C1, C1,  C1,  9);
  packw(b2_c1_w, wp_b2c1,C2, C1,  C1,  9);
  packw(b2_sc_w, wp_b2sc,C2, C1,  C1,  1);
  packw(b2_c2_w, wp_b2c2,C2, C2,  C2,  9);
  packw(conv3_w, wp_c3,  16, C2,  C2,  1);

  auto conv = [&](const f16* yin, int Hin, int Win, int Cpad, const f16* wp, int Cout,
                  int Hout, int Wout, int stride, int pad, int KW, int taps,
                  const float* addsrc, const float* bias, float* o) {
    int M = BB * Hout * Wout;
    dim3 grid(M / 16, (Cout + 63) / 64);
    k_conv<<<grid, 128, 0, stream>>>(yin, Hin, Win, Cpad, wp, Cout, Hout, Wout,
                                     stride, pad, KW, taps, addsrc, bias, o);
  };
  auto bnrelu = [&](const float* in, const float* g, const float* bb, f16* o, int Npix, int C, int Cpad) {
    int n = Npix * Cpad;
    k_bnrelu<<<(n + 255) / 256, 256, 0, stream>>>(in, g, bb, o, Npix, C, Cpad);
  };

  // ---- block 1 (stride 2): 517ch 32x42 -> 256ch 16x21
  conv(y1, HH, WW, CP1, wp_sc1, C1, H1, W1, 2, 0, 1, 1, nullptr, nullptr, sc1out);
  conv(y1, HH, WW, CP1, wp_c1,  C1, H1, W1, 2, 1, 3, 9, nullptr, nullptr, c1out);
  bnrelu(c1out, b1_bn2_g, b1_bn2_b, z1, BB * HW1, C1, C1);
  conv(z1, H1, W1, C1, wp_c2, C1, H1, W1, 1, 1, 3, 9, sc1out, nullptr, blk1);

  // ---- block 2 (stride 2): 256ch 16x21 -> 128ch 8x11
  bnrelu(blk1, b2_bn1_g, b2_bn1_b, y2, BB * HW1, C1, C1);
  conv(y2, H1, W1, C1, wp_b2sc, C2, H2, W2, 2, 0, 1, 1, nullptr, nullptr, sc2out);
  conv(y2, H1, W1, C1, wp_b2c1, C2, H2, W2, 2, 1, 3, 9, nullptr, nullptr, c2out);
  bnrelu(c2out, b2_bn2_g, b2_bn2_b, z2, BB * HW2, C2, C2);
  conv(z2, H2, W2, C2, wp_b2c2, C2, H2, W2, 1, 1, 3, 9, sc2out, nullptr, blk2);

  // ---- bn3 + relu + 1x1 conv3 (+bias)
  bnrelu(blk2, bn3_g, bn3_b, y3, BB * HW2, C2, C2);
  conv(y3, H2, W2, C2, wp_c3, 16, H2, W2, 1, 0, 1, 1, nullptr, conv3_b, c3out);

  // ---- head MLP + quaternion
  k_head1<<<(BB * 128 + 255) / 256, 256, 0, stream>>>(c3out, head_w1, head_b1, hdd);
  k_head2<<<1, 32, 0, stream>>>(hdd, head_w2, head_b2, out);
  (void)out_size; (void)n_in; (void)in_sizes;
}
